// NeuralMemory_36953898615485
// MI455X (gfx1250) — matrix operand order repaired and checked by reading, compiled, not run on hardware
//
#include <hip/hip_runtime.h>

typedef __bf16 bf16;
typedef __attribute__((ext_vector_type(16))) __bf16 v16bf;
typedef __attribute__((ext_vector_type(8)))  __bf16 v8bf;
typedef __attribute__((ext_vector_type(8)))  float   v8f;

#define DIM    512
#define NROWS  65536
#define BROWS  64
#define LSTR   520          // padded LDS row stride in bf16 elems (1040 B, 16B-aligned)
#define TPB1   256          // 8 waves
#define SPLITK 64
#define KCHUNK (NROWS / SPLITK)

union ABu { v16bf v; v8bf h[2]; };

// A-fragment (16x32 bf16, M x K) from LDS row-major tile.
// lane: m = lane&15, khalf = lane>>4; elem e: K = khalf*16 + e
__device__ __forceinline__ v16bf load_afrag(const bf16* lds, int m, int kk, int khalf) {
    ABu u;
    u.h[0] = *(const v8bf*)(lds + m * LSTR + kk + khalf * 16);
    u.h[1] = *(const v8bf*)(lds + m * LSTR + kk + khalf * 16 + 8);
    return u.v;
}

// GEMM stage: Y[64x512] = act(X[64x512] @ Wb^T + bias).
// 8 waves; each wave owns a 64(row) x 64(col) block = 4x4 WMMA tiles,
// reusing every A and B fragment 4x (L2 bytes/WMMA cut 4x vs 1x8 blocking).
template<bool SILU, bool TO_LDS, bool TO_GT, bool DIFF, bool TO_OUT>
__device__ __forceinline__ void gemm_stage(
    const bf16* __restrict__ Wb, const float* __restrict__ bias,
    const bf16* __restrict__ ldsX, bf16* __restrict__ ldsY,
    const bf16* __restrict__ ldsV, bf16* __restrict__ gT,
    float* __restrict__ outF, int rowGlobalBase)
{
    const int tid   = threadIdx.x;
    const int wave  = tid >> 5;           // 0..7
    const int lane  = tid & 31;
    const int khalf = lane >> 4;
    const int l16   = lane & 15;
    const int colBase = wave * 64;

    v8f acc[4][4] = {};
    for (int kk = 0; kk < DIM; kk += 32) {
        v16bf a[4];
#pragma unroll
        for (int rt = 0; rt < 4; rt++)
            a[rt] = load_afrag(ldsX, rt * 16 + l16, kk, khalf);
#pragma unroll
        for (int t = 0; t < 4; t++) {
            const int n = colBase + t * 16 + l16;
            // B-fragment (32x16, K x N): lane n = lane&15, K = khalf*16 + e (contig 32B)
            v16bf b = *(const v16bf*)(Wb + (size_t)n * DIM + kk + khalf * 16);
#pragma unroll
            for (int rt = 0; rt < 4; rt++)
                acc[rt][t] = __builtin_amdgcn_wmma_f32_16x16x32_bf16(
                    false, a[rt], false, b, (short)0, acc[rt][t], false, false);
        }
    }
#pragma unroll
    for (int t = 0; t < 4; t++) {
        const int col = colBase + t * 16 + l16;
        const float bv = bias[col];
#pragma unroll
        for (int rt = 0; rt < 4; rt++) {
            const int rowLocal = rt * 16 + khalf * 8;
            float vals[8];
#pragma unroll
            for (int r = 0; r < 8; r++) {
                float x = acc[rt][t][r] + bv;
                if constexpr (SILU) x = x / (1.0f + __expf(-x));
                vals[r] = x;
            }
            if constexpr (DIFF) {
#pragma unroll
                for (int r = 0; r < 8; r++)
                    vals[r] -= (float)ldsV[(rowLocal + r) * LSTR + col];
            }
            if constexpr (TO_LDS) {
#pragma unroll
                for (int r = 0; r < 8; r++)
                    ldsY[(rowLocal + r) * LSTR + col] = (bf16)vals[r];
            }
            if constexpr (TO_GT) {  // transposed: gT[col][rowGlobal], 8 contig rows = 16B store
                v8bf pk;
#pragma unroll
                for (int r = 0; r < 8; r++) pk[r] = (bf16)vals[r];
                *(v8bf*)(gT + (size_t)col * NROWS + rowGlobalBase + rowLocal) = pk;
            }
            if constexpr (TO_OUT) {
#pragma unroll
                for (int r = 0; r < 8; r++)
                    outF[(size_t)(rowGlobalBase + rowLocal + r) * DIM + col] = vals[r];
            }
        }
    }
}

__device__ __forceinline__ void norm_rows(const float* __restrict__ src,
                                          bf16* __restrict__ dst, int rowBase)
{
    const int tid = threadIdx.x, wave = tid >> 5, lane = tid & 31;
#pragma unroll
    for (int rr = 0; rr < 8; rr++) {
        const int row = wave * 8 + rr;
        const float* p = src + (size_t)(rowBase + row) * DIM;
        float x[16]; float ss = 0.0f;
#pragma unroll
        for (int i = 0; i < 16; i++) { x[i] = p[lane + i * 32]; ss += x[i] * x[i]; }
#pragma unroll
        for (int off = 16; off >= 1; off >>= 1) ss += __shfl_xor(ss, off, 32);
        const float inv = 1.0f / fmaxf(sqrtf(ss), 1e-12f);
#pragma unroll
        for (int i = 0; i < 16; i++) dst[row * LSTR + lane + i * 32] = (bf16)(x[i] * inv);
    }
}

__global__ void prep_kernel(const float* inW, const float* l0W, const float* l1W,
                            const float* memW, const float* outW,
                            bf16* wb, float* grad)
{
    const int idx = blockIdx.x * 256 + threadIdx.x;
    if (idx < DIM * DIM) {
        wb[0 * DIM * DIM + idx] = (bf16)inW[idx];
        wb[1 * DIM * DIM + idx] = (bf16)l0W[idx];
        wb[2 * DIM * DIM + idx] = (bf16)l1W[idx];
        wb[3 * DIM * DIM + idx] = (bf16)memW[idx];
        wb[4 * DIM * DIM + idx] = (bf16)outW[idx];
        grad[idx] = 0.0f;
    }
}

__global__ __launch_bounds__(TPB1)
void fwd_kernel(const float* __restrict__ key, const float* __restrict__ value,
                const bf16* __restrict__ inWb, const float* __restrict__ in_b,
                const bf16* __restrict__ l0Wb, const float* __restrict__ l0_b,
                const bf16* __restrict__ l1Wb, const float* __restrict__ l1_b,
                const bf16* __restrict__ memWb, const float* __restrict__ mem_b,
                bf16* __restrict__ k3, bf16* __restrict__ k3T, bf16* __restrict__ diffT)
{
    extern __shared__ bf16 smem[];
    bf16* ldsX = smem;
    bf16* ldsY = smem + BROWS * LSTR;
    bf16* ldsV = smem + 2 * BROWS * LSTR;
    const int rowBase = blockIdx.x * BROWS;
    const int tid = threadIdx.x;

    // v_proj = l2norm(value) @ in_W^T + in_b  -> ldsV
    norm_rows(value, ldsX, rowBase);
    __syncthreads();
    gemm_stage<false, true, false, false, false>(inWb, in_b, ldsX, ldsV, nullptr, nullptr, nullptr, rowBase);
    __syncthreads();
    // k path
    norm_rows(key, ldsX, rowBase);
    __syncthreads();
    gemm_stage<false, true, false, false, false>(inWb, in_b, ldsX, ldsY, nullptr, nullptr, nullptr, rowBase);
    __syncthreads();
    gemm_stage<true, true, false, false, false>(l0Wb, l0_b, ldsY, ldsX, nullptr, nullptr, nullptr, rowBase);
    __syncthreads();
    // k3 = silu(... @ l1_W^T + l1_b)  -> ldsY and k3T (transposed, for grad GEMM)
    gemm_stage<true, true, true, false, false>(l1Wb, l1_b, ldsX, ldsY, nullptr, k3T, nullptr, rowBase);
    __syncthreads();
    // coalesced row-major k3 spill (for the final kernel)
    for (int idx = tid; idx < BROWS * (DIM / 8); idx += TPB1) {
        const int row = idx >> 6, c8 = idx & 63;
        uint4 v = *(const uint4*)(ldsY + row * LSTR + c8 * 8);
        *(uint4*)(k3 + (size_t)(rowBase + row) * DIM + c8 * 8) = v;
    }
    // diff = (k3 @ mem_W^T + mem_b) - v_proj -> diffT (transposed)
    gemm_stage<false, false, true, true, false>(memWb, mem_b, ldsY, nullptr, ldsV, diffT, nullptr, rowBase);
}

// grad += diffT[512 x N] x k3T[512 x N]^T  (reduction over N, split-K + f32 atomics).
// 8 waves/block; each wave owns 32x32 = 2x2 tiles (A and B fragments reused 2x).
__global__ __launch_bounds__(256)
void grad_kernel(const bf16* __restrict__ diffT, const bf16* __restrict__ k3T,
                 float* __restrict__ grad)
{
    const int region = blockIdx.x;        // 0..31 : 8 m-regions(64) x 4 n-regions(128)
    const int split  = blockIdx.y;        // 0..SPLITK-1
    const int tid = threadIdx.x, wave = tid >> 5, lane = tid & 31;
    const int khalf = lane >> 4, l16 = lane & 15;
    const int mBase = (region & 7) * 64 + (wave & 1) * 32;
    const int nBase = (region >> 3) * 128 + (wave >> 1) * 32;
    const size_t krow0 = (size_t)split * KCHUNK;

    const bf16* arow0 = diffT + (size_t)(mBase + l16) * NROWS + krow0;
    const bf16* arow1 = arow0 + (size_t)16 * NROWS;
    const bf16* brow0 = k3T + (size_t)(nBase + l16) * NROWS + krow0;
    const bf16* brow1 = brow0 + (size_t)16 * NROWS;

    v8f acc[2][2] = {};
    for (int kk = 0; kk < KCHUNK; kk += 32) {
        ABu a[2];
        a[0].h[0] = *(const v8bf*)(arow0 + kk + khalf * 16);
        a[0].h[1] = *(const v8bf*)(arow0 + kk + khalf * 16 + 8);
        a[1].h[0] = *(const v8bf*)(arow1 + kk + khalf * 16);
        a[1].h[1] = *(const v8bf*)(arow1 + kk + khalf * 16 + 8);
        v16bf b0 = *(const v16bf*)(brow0 + kk + khalf * 16);
        v16bf b1 = *(const v16bf*)(brow1 + kk + khalf * 16);
#pragma unroll
        for (int i = 0; i < 2; i++) {
            acc[i][0] = __builtin_amdgcn_wmma_f32_16x16x32_bf16(
                false, a[i].v, false, b0, (short)0, acc[i][0], false, false);
            acc[i][1] = __builtin_amdgcn_wmma_f32_16x16x32_bf16(
                false, a[i].v, false, b1, (short)0, acc[i][1], false, false);
        }
    }
#pragma unroll
    for (int i = 0; i < 2; i++) {
#pragma unroll
        for (int j = 0; j < 2; j++) {
#pragma unroll
            for (int r = 0; r < 8; r++) {
                const int m = mBase + i * 16 + khalf * 8 + r;
                const int n = nBase + j * 16 + l16;
                __hip_atomic_fetch_add(&grad[m * DIM + n], acc[i][j][r],
                                       __ATOMIC_RELAXED, __HIP_MEMORY_SCOPE_AGENT);
            }
        }
    }
}

__global__ void update_kernel(const float* __restrict__ memW, const float* __restrict__ grad,
                              const float* __restrict__ forget, const float* __restrict__ lr,
                              bf16* __restrict__ uWb)
{
    const int idx = blockIdx.x * 256 + threadIdx.x;
    if (idx < DIM * DIM) {
        const float scale = 2.0f / ((float)NROWS * (float)DIM);
        const float u = (1.0f - forget[0]) * memW[idx] + lr[0] * (grad[idx] * scale);
        uWb[idx] = (bf16)u;
    }
}

__global__ __launch_bounds__(TPB1)
void out_kernel(const bf16* __restrict__ k3, const bf16* __restrict__ uWb,
                const float* __restrict__ mem_b, const bf16* __restrict__ outWb,
                const float* __restrict__ out_b, float* __restrict__ out)
{
    extern __shared__ bf16 smem[];
    bf16* ldsX = smem;
    bf16* ldsY = smem + BROWS * LSTR;
    const int rowBase = blockIdx.x * BROWS;
    const int tid = threadIdx.x;

    for (int idx = tid; idx < BROWS * (DIM / 8); idx += TPB1) {
        const int row = idx >> 6, c8 = idx & 63;
        uint4 v = *(const uint4*)(k3 + (size_t)(rowBase + row) * DIM + c8 * 8);
        *(uint4*)(ldsX + row * LSTR + c8 * 8) = v;
    }
    __syncthreads();
    gemm_stage<false, true, false, false, false>(uWb, mem_b, ldsX, ldsY, nullptr, nullptr, nullptr, rowBase);
    __syncthreads();
    gemm_stage<false, false, false, false, true>(outWb, out_b, ldsY, nullptr, nullptr, nullptr, out, rowBase);
}

extern "C" void kernel_launch(void* const* d_in, const int* in_sizes, int n_in,
                              void* d_out, int out_size, void* d_ws, size_t ws_size,
                              hipStream_t stream) {
    (void)in_sizes; (void)n_in; (void)out_size; (void)ws_size;
    const float* key    = (const float*)d_in[0];
    const float* value  = (const float*)d_in[1];
    const float* in_W   = (const float*)d_in[2];
    const float* in_b   = (const float*)d_in[3];
    const float* l0_W   = (const float*)d_in[4];
    const float* l0_b   = (const float*)d_in[5];
    const float* l1_W   = (const float*)d_in[6];
    const float* l1_b   = (const float*)d_in[7];
    const float* mem_W  = (const float*)d_in[8];
    const float* mem_b  = (const float*)d_in[9];
    const float* out_W  = (const float*)d_in[10];
    const float* out_b  = (const float*)d_in[11];
    const float* forget = (const float*)d_in[12];
    const float* lr     = (const float*)d_in[13];
    float* out = (float*)d_out;

    // workspace layout
    bf16*  wb    = (bf16*)d_ws;                       // 5 x [512x512] bf16
    bf16*  inWb  = wb + 0 * DIM * DIM;
    bf16*  l0Wb  = wb + 1 * DIM * DIM;
    bf16*  l1Wb  = wb + 2 * DIM * DIM;
    bf16*  memWb = wb + 3 * DIM * DIM;
    bf16*  outWb = wb + 4 * DIM * DIM;
    bf16*  uWb   = wb + 5 * DIM * DIM;                // updated mem_W, bf16
    float* grad  = (float*)(wb + 6 * DIM * DIM);      // [512x512] f32
    bf16*  k3    = (bf16*)(grad + DIM * DIM);         // [N x 512] row-major
    bf16*  k3T   = k3 + (size_t)NROWS * DIM;          // [512 x N]
    bf16*  diffT = k3T + (size_t)NROWS * DIM;         // [512 x N]

    prep_kernel<<<(DIM * DIM) / 256, 256, 0, stream>>>(in_W, l0_W, l1_W, mem_W, out_W, wb, grad);

    const size_t lds1 = (size_t)3 * BROWS * LSTR * sizeof(bf16);
    fwd_kernel<<<NROWS / BROWS, TPB1, lds1, stream>>>(
        key, value, inWb, in_b, l0Wb, l0_b, l1Wb, l1_b, memWb, mem_b, k3, k3T, diffT);

    dim3 gg(32, SPLITK);
    grad_kernel<<<gg, 256, 0, stream>>>(diffT, k3T, grad);

    update_kernel<<<(DIM * DIM) / 256, 256, 0, stream>>>(mem_W, grad, forget, lr, uWb);

    const size_t lds4 = (size_t)2 * BROWS * LSTR * sizeof(bf16);
    out_kernel<<<NROWS / BROWS, TPB1, lds4, stream>>>(k3, uWb, mem_b, outWb, out_b, out);
}
